// LSTM_74113955659941
// MI455X (gfx1250) — matrix operand order repaired and checked by reading, not compiled, not run on hardware
//
#include <hip/hip_runtime.h>
#include <hip/hip_bf16.h>

typedef __attribute__((ext_vector_type(16))) _Float16 v16h;
typedef __attribute__((ext_vector_type(8)))  _Float16 v8h;
typedef __attribute__((ext_vector_type(8)))  float    v8f;

namespace {
constexpr int kH      = 27;        // hidden = input = classes
constexpr int kG      = 108;       // 4*H gate rows
constexpr int kT      = 78;        // timesteps
constexpr int kNWaves = 4;         // waves per block
constexpr int kRows   = 16;        // batch rows per wave = WMMA M
constexpr int kNFrag  = 50;        // 3 layers * 2 mats * 8 gate tiles + 2 head
constexpr int kTH     = kT * kH;   // 2106: batch-row stride of x/out (elements)
}

__device__ __forceinline__ float tanh_fast(float x) {
#if __has_builtin(__builtin_amdgcn_tanhf)
  return __builtin_amdgcn_tanhf(x);           // gfx1250 hardware v_tanh_f32
#else
  float r;
  asm("v_tanh_f32 %0, %1" : "=v"(r) : "v"(x));
  return r;
#endif
}
__device__ __forceinline__ float sigmoid_fast(float x) {
  return 0.5f * tanh_fast(0.5f * x) + 0.5f;   // 1 trans-op sigmoid
}

// Compile-time scheduling fence: stops the scheduler from hoisting the next
// layer's independent recurrent WMMAs / fragment loads into this region,
// which otherwise triples live accumulators and pushes allocation past 256
// VGPRs (s_set_vgpr_msb storm seen in round 3).
__device__ __forceinline__ void sched_fence() {
#if __has_builtin(__builtin_amdgcn_sched_barrier)
  __builtin_amdgcn_sched_barrier(0);
#endif
}

__device__ __forceinline__ v8f wmma_f16(v16h a, v16h b, v8f c) {
  return __builtin_amdgcn_wmma_f32_16x16x32_f16(false, a, false, b,
                                                (short)0, c, false, false);
}

// A-fragment (16x32 f16) from [16 rows][32 cols] f16 LDS tile.
// lane (nl,kh): elems 0..7 = K[kh*8..+7] of row nl, elems 8..15 = K[16+kh*8..+7].
__device__ __forceinline__ v16h load_afrag(const _Float16* b, int nl, int kh) {
  v8h lo = *(const v8h*)(b + nl * 32 + kh * 8);
  v8h hi = *(const v8h*)(b + nl * 32 + kh * 8 + 16);
  v16h a;
#pragma unroll
  for (int e = 0; e < 8; ++e) { a[e] = lo[e]; a[8 + e] = hi[e]; }
  return a;
}

// B-fragment from split lo/hi LDS halves (16B lane stride = conflict-free).
__device__ __forceinline__ v16h load_bfrag(const _Float16* wfA,
                                           const _Float16* wfB,
                                           int f, int lane) {
  v8h lo = *(const v8h*)(wfA + (f * 32 + lane) * 8);
  v8h hi = *(const v8h*)(wfB + (f * 32 + lane) * 8);
  v16h b;
#pragma unroll
  for (int e = 0; e < 8; ++e) { b[e] = lo[e]; b[8 + e] = hi[e]; }
  return b;
}

// Load one x row's A-fragment K-values straight from global (row-major, so the
// fragment's K runs are contiguous): lo = K[kh*8..+7] (always valid),
// hi = K[16+kh*8..+7] with K>=27 zeroed (guards are compile-time per branch).
__device__ __forceinline__ void load_xrow(const float* xrow, int kh,
                                          float* lo, float* hi) {
#pragma unroll
  for (int e = 0; e < 8; ++e)
    lo[e] = __builtin_nontemporal_load(xrow + kh * 8 + e);
  if (kh == 0) {
#pragma unroll
    for (int e = 0; e < 8; ++e)
      hi[e] = __builtin_nontemporal_load(xrow + 16 + e);
  } else {
#pragma unroll
    for (int e = 0; e < 8; ++e)
      hi[e] = (24 + e < kH) ? __builtin_nontemporal_load(xrow + 24 + e) : 0.0f;
  }
}
__device__ __forceinline__ v16h pack_afrag(const float* lo, const float* hi) {
  v16h a;
#pragma unroll
  for (int e = 0; e < 8; ++e) {
    a[e]     = (_Float16)lo[e];
    a[8 + e] = (_Float16)hi[e];
  }
  return a;
}

__global__ __launch_bounds__(kNWaves * 32)
void lstm3_fused(const float* __restrict__ x, const float* __restrict__ h0,
                 const float* __restrict__ c0, const float* __restrict__ Wih,
                 const float* __restrict__ Whh, const float* __restrict__ bih,
                 const float* __restrict__ bhh, const float* __restrict__ Wl,
                 const float* __restrict__ bl, float* __restrict__ out, int B) {
  __shared__ _Float16 wfA[kNFrag * 32 * 8];        // frag elems 0..7
  __shared__ _Float16 wfB[kNFrag * 32 * 8];        // frag elems 8..15
  __shared__ _Float16 hbuf[kNWaves][kRows * 32];   // h staging (f16, K-padded)
  // LDS: 25600 + 25600 + 4096 = 55296 B

  const int tid  = threadIdx.x;
  const int lane = tid & 31;
  const int w    = tid >> 5;
  const int nl   = lane & 15;   // B/C-frag column within tile
  const int kh   = lane >> 4;   // lane-half

  // ---- build gate-aligned pre-swizzled f16 B-fragments ----
  // Gate layout padded to 4 gates x 32 cols: tile (c,g) covers col h=16c+nl of
  // gate g, i.e. weight row g*27+h.  frag id: ((l*2+mat)*2+c)*4+g ; head 48+j.
  for (int f = w; f < kNFrag; f += kNWaves) {
    const float* W;
    int row, valid_n;
    if (f < 48) {
      const int g = f & 3, c = (f >> 2) & 1, mat = (f >> 3) & 1, l = f >> 4;
      const int h = 16 * c + nl;
      W = (mat ? Whh : Wih) + l * kG * kH;
      row = g * kH + h;
      valid_n = (h < kH);
    } else {
      const int j = f - 48;                 // head: plain col tiles of Wl
      W = Wl;
      row = 16 * j + nl;
      valid_n = (row < kH);
    }
    _Float16* da = &wfA[(f * 32 + lane) * 8];
    _Float16* db = &wfB[(f * 32 + lane) * 8];
#pragma unroll
    for (int e = 0; e < 8; ++e) {
      const int ka = kh * 8 + e, kb = 16 + kh * 8 + e;
      da[e] = (valid_n && ka < kH) ? (_Float16)W[row * kH + ka] : (_Float16)0.0f;
      db[e] = (valid_n && kb < kH) ? (_Float16)W[row * kH + kb] : (_Float16)0.0f;
    }
  }
  for (int i = lane; i < kRows * 32; i += 32) hbuf[w][i] = (_Float16)0.0f;
  __syncthreads();

  const int rowbase = blockIdx.x * (kNWaves * kRows) + w * kRows;
  const int rlo     = rowbase + 8 * kh;     // this lane's first batch row

  // ---- per-lane biases (the only gate columns this lane ever touches) ----
  float bg[3][2][4], blr[2];
#pragma unroll
  for (int l = 0; l < 3; ++l)
#pragma unroll
    for (int c = 0; c < 2; ++c) {
      const int h = 16 * c + nl;
#pragma unroll
      for (int g = 0; g < 4; ++g)
        bg[l][c][g] = (h < kH) ? (bih[l * kG + g * kH + h] +
                                  bhh[l * kG + g * kH + h]) : 0.0f;
    }
#pragma unroll
  for (int j = 0; j < 2; ++j)
    blr[j] = (16 * j + nl < kH) ? bl[16 * j + nl] : 0.0f;

  // ---- state: c in fp32 regs (8 rows x 2 cols per lane), h as A-fragments ----
  float creg[3][16];
  v16h  ah[3];
#pragma unroll
  for (int l = 0; l < 3; ++l) {
#pragma unroll
    for (int c = 0; c < 2; ++c) {
      const int h = 16 * c + nl;
#pragma unroll
      for (int r = 0; r < 8; ++r) {
        creg[l][c * 8 + r] =
            (h < kH) ? c0[((size_t)l * B + rlo + r) * kH + h] : 0.0f;
        if (h < kH)
          hbuf[w][(8 * kh + r) * 32 + h] =
              (_Float16)h0[((size_t)l * B + rlo + r) * kH + h];
      }
    }
    ah[l] = load_afrag(hbuf[w], nl, kh);
  }

  // prefetch x_t=0
  const float* xrow = x + (size_t)(rowbase + nl) * kTH;
  float xlo[8], xhi[8];
  load_xrow(xrow, kh, xlo, xhi);

  // ---- time loop: 3 layers + head fused per step ----
  for (int t = 0; t < kT; ++t) {
    v16h ain = pack_afrag(xlo, xhi);
    if (t + 1 < kT) load_xrow(xrow + (size_t)(t + 1) * kH, kh, xlo, xhi);

#pragma unroll
    for (int l = 0; l < 3; ++l) {
#pragma unroll
      for (int c = 0; c < 2; ++c) {
        v8f ac[4];
#pragma unroll
        for (int g = 0; g < 4; ++g) {
          const int fb = ((l * 2 + 0) * 2 + c) * 4 + g;   // Wih frag
          const int fu = ((l * 2 + 1) * 2 + c) * 4 + g;   // Whh frag
          const v16h bw = load_bfrag(wfA, wfB, fb, lane);
          const v16h bu = load_bfrag(wfA, wfB, fu, lane);
          v8f acc = {};
          acc = wmma_f16(ain, bw, acc);
          acc = wmma_f16(ah[l], bu, acc);
          ac[g] = acc;
        }
        // cell update fully in registers: this lane owns rows rlo..rlo+7,
        // gate column h = 16c+nl, all four gates in ac[0..3].
        const int h = 16 * c + nl;
        if (h < kH) {
          _Float16* hp = &hbuf[w][(8 * kh) * 32 + h];
#pragma unroll
          for (int r = 0; r < 8; ++r) {
            const float gi = sigmoid_fast(ac[0][r] + bg[l][c][0]);
            const float gf = sigmoid_fast(ac[1][r] + bg[l][c][1]);
            const float gg = tanh_fast  (ac[2][r] + bg[l][c][2]);
            const float go = sigmoid_fast(ac[3][r] + bg[l][c][3]);
            const float cc = gf * creg[l][c * 8 + r] + gi * gg;
            creg[l][c * 8 + r] = cc;
            hp[r * 32] = (_Float16)(go * tanh_fast(cc));
          }
        }
        sched_fence();   // bound liveness to one gate-column group
      }
      ain   = load_afrag(hbuf[w], nl, kh);   // h_t^l -> layer l+1 input ...
      ah[l] = ain;                           // ... and recurrent input at t+1
      sched_fence();     // keep layers' independent WMMAs from interleaving
    }

    // head: y = h3 @ Wl^T + bl, stored straight from accumulators
    float* orow = out + (size_t)t * kH + (size_t)(rowbase + 8 * kh) * kTH + nl;
#pragma unroll
    for (int j = 0; j < 2; ++j) {
      const v16h bwl = load_bfrag(wfA, wfB, 48 + j, lane);
      v8f acc = {};
      acc = wmma_f16(ah[2], bwl, acc);
      if (16 * j + nl < kH) {
#pragma unroll
        for (int r = 0; r < 8; ++r)
          __builtin_nontemporal_store(acc[r] + blr[j],
                                      orow + (size_t)r * kTH + 16 * j);
      }
    }
    sched_fence();
  }
}

extern "C" void kernel_launch(void* const* d_in, const int* in_sizes, int n_in,
                              void* d_out, int out_size, void* d_ws, size_t ws_size,
                              hipStream_t stream) {
  (void)n_in; (void)out_size; (void)d_ws; (void)ws_size;
  const float* x   = (const float*)d_in[0];
  const float* h0  = (const float*)d_in[1];
  const float* c0  = (const float*)d_in[2];
  const float* Wih = (const float*)d_in[3];
  const float* Whh = (const float*)d_in[4];
  const float* bih = (const float*)d_in[5];
  const float* bhh = (const float*)d_in[6];
  const float* Wl  = (const float*)d_in[7];
  const float* bl  = (const float*)d_in[8];
  float* out = (float*)d_out;

  const int B = in_sizes[0] / (kT * kH);            // 32768
  const int grid = B / (kNWaves * kRows);           // 512 blocks of 128 threads
  lstm3_fused<<<grid, kNWaves * 32, 0, stream>>>(x, h0, c0, Wih, Whh, bih, bhh,
                                                 Wl, bl, out, B);
}
